// Attention_32547262169382
// MI455X (gfx1250) — matrix operand order repaired and checked by reading, not compiled, not run on hardware
//
#include <hip/hip_runtime.h>
#include <math.h>

// ---------------------------------------------------------------------------
// Causal attention (B=4, S=2048, D=1024, fp32) for MI455X / gfx1250.
//
// Matrix math: V_WMMA_F32_16X16X32_BF16 with "bf16 x 3" split operands
// (hi/lo decomposition -> ~17 mantissa bits/operand, fp32 accumulate,
//  rel err ~1e-4 over K=1024 -- fp32-like accuracy at 8x the MACs/instr of
//  the fp32 16x16x4 WMMA path).
//
// Data movement: Tensor Data Mover (TENSOR_LOAD_TO_LDS). Every GEMM operand
// is stored contiguous-in-k ([M,K] for A, [N,K] for B), so each 64x32 / 128x32
// tile is one 2-D TDM descriptor. TDM applies LDS padding in flight
// (pad 16B after every 64B row -> 80B pitch = bank-conflict-free b128 frag
// reads). Double-buffered: wave 0 posts tile i+1 while all waves compute i;
// completion tracked with TENSORcnt (s_wait_tensorcnt).
//
// Pipeline (intermediates in d_ws):
//   0) split x -> hi/lo bf16; transpose+split Wq/Wk/Wv/Wo -> [N,K] hi/lo
//   1) Q/K = x @ W + b -> hi/lo bf16 ; V -> hi/lo bf16 stored transposed [b,d,s]
//   2) scores = Q @ K^T / 8 -> fp32 (skip upper-triangular blocks)
//   3) causal softmax -> P hi/lo bf16
//   4) attn = P @ V (causal k-limit) -> hi/lo bf16
//   5) out = attn @ Wo + bo -> fp32
// ---------------------------------------------------------------------------

typedef __bf16 v16bf __attribute__((ext_vector_type(16)));
typedef float v8f __attribute__((ext_vector_type(8)));
typedef unsigned int v4u __attribute__((ext_vector_type(4)));
typedef float v4f __attribute__((ext_vector_type(4)));
typedef unsigned int v4ui __attribute__((ext_vector_type(4)));
typedef int v4si __attribute__((ext_vector_type(4)));
typedef int v8si __attribute__((ext_vector_type(8)));
typedef unsigned short u16;

#define LP 40  // LDS pitch in ushorts: 64B row + 16B TDM pad = 80B (16B aligned)

__device__ __forceinline__ u16 bf16_rne(float f) {
  unsigned u = __builtin_bit_cast(unsigned, f);
  unsigned r = u + 0x7fffu + ((u >> 16) & 1u);
  return (u16)(r >> 16);
}
__device__ __forceinline__ float bf16_to_f(u16 h) {
  unsigned u = ((unsigned)h) << 16;
  return __builtin_bit_cast(float, u);
}

struct Pack32 { v4u a, b; };
__device__ __forceinline__ v16bf frag_ld(const u16* p0, const u16* p1) {
  Pack32 t;
  t.a = *(const v4u*)p0;
  t.b = *(const v4u*)p1;
  return __builtin_bit_cast(v16bf, t);
}

// Workgroup-relative LDS byte offset of a __shared__ object (flat-pointer LSBs).
__device__ __forceinline__ unsigned lds_off(const void* p) {
  return (unsigned)(unsigned long long)p;
}

// Issue one TDM 2-D tile load: `rows` x 32 bf16 elements from row-major
// (leading dim `ld` elements) starting at gsrc, into LDS at dst_off with
// 16B padding after each 64B row (-> LP pitch).  D# per ISA ch.8.
__device__ __forceinline__ void tdm_tile_2d(const u16* gsrc, int ld, int rows,
                                            unsigned dst_off) {
  const unsigned long long ga = (unsigned long long)gsrc;
  v4ui g0;
  g0[0] = 1u;                                    // count=1 (valid user D#)
  g0[1] = dst_off;                               // lds_addr (bytes)
  g0[2] = (unsigned)(ga & 0xffffffffu);          // global_addr[31:0]
  g0[3] = (unsigned)((ga >> 32) & 0x01ffffffu)   // global_addr[56:32]
          | (2u << 30);                          // type=2 ("image")
  v8si g1;
  g1[0] = (int)((1u << 16)                       // data_size: 2 bytes
                | (1u << 20)                     // pad_enable
                | (3u << 22)                     // pad_interval: 16 DWORDs (64B)
                | (3u << 25));                   // pad_amount: 4 DWORDs (16B)
  g1[1] = (int)((((unsigned)ld) & 0xffffu) << 16);          // tensor_dim0[15:0]
  g1[2] = (int)(((((unsigned)ld) >> 16) & 0xffffu)          // tensor_dim0[31:16]
                | (((unsigned)rows & 0xffffu) << 16));      // tensor_dim1[15:0]
  g1[3] = (int)(32u << 16);                      // tensor_dim1 hi=0 | tile_dim0=32
  g1[4] = (int)((unsigned)rows & 0xffffu);       // tile_dim1=rows, tile_dim2=0
  g1[5] = (int)(unsigned)ld;                     // tensor_dim0_stride[31:0]
  g1[6] = 0;                                     // stride hi / dim1_stride lo
  g1[7] = 0;
  const v4si zz = (v4si){0, 0, 0, 0};
#if __clang_major__ >= 23
  const v8si z8 = (v8si){0, 0, 0, 0, 0, 0, 0, 0};
  __builtin_amdgcn_tensor_load_to_lds(g0, g1, zz, zz, z8, 0);
#else
  __builtin_amdgcn_tensor_load_to_lds(g0, g1, zz, zz, 0);
#endif
}

// Elementwise split: hi = bf16(v), lo = bf16(v - hi). n multiple of 1024.
__global__ __launch_bounds__(256) void split_bf16(const float* __restrict__ in,
                                                  u16* __restrict__ hi,
                                                  u16* __restrict__ lo, long long n) {
  long long i = ((long long)blockIdx.x * 256 + threadIdx.x) * 4;
  v4f v = *(const v4f*)(in + i);
#pragma unroll
  for (int j = 0; j < 4; j++) {
    u16 h = bf16_rne(v[j]);
    hi[i + j] = h;
    lo[i + j] = bf16_rne(v[j] - bf16_to_f(h));
  }
}

// Transpose + split a 1024x1024 fp32 weight: out[n][k] = in[k][n], hi/lo bf16.
__global__ __launch_bounds__(256) void transpose_split(const float* __restrict__ in,
                                                       u16* __restrict__ outh,
                                                       u16* __restrict__ outl) {
  __shared__ float t[32][33];
  const int k0 = blockIdx.x * 32, n0 = blockIdx.y * 32;
  const int tx = threadIdx.x, ty = threadIdx.y;  // block (32, 8)
#pragma unroll
  for (int j = 0; j < 4; j++)
    t[ty + j * 8][tx] = in[(long long)(k0 + ty + j * 8) * 1024 + n0 + tx];
  __syncthreads();
#pragma unroll
  for (int j = 0; j < 4; j++) {
    const float v = t[tx][ty + j * 8];
    const long long idx = (long long)(n0 + ty + j * 8) * 1024 + k0 + tx;
    const u16 h = bf16_rne(v);
    outh[idx] = h;
    outl[idx] = bf16_rne(v - bf16_to_f(h));
  }
}

// bf16x3 GEMM: C = alpha * A(MxK) * B^T + bias, with A [M,K] and B [N,K]
// (both contiguous in k). Block tile 64(M) x 128(N), 256 threads = 8 waves,
// each wave a 16x64 strip (4 accumulators). Tiles fetched by TDM, double-buffered.
//   CAUSAL_K:   limit k-loop to k < m0+64
//   SKIP_UPPER: skip blocks fully above the diagonal
//   OUT_MODE:   0 = fp32, 1 = hi/lo bf16, 2 = hi/lo bf16 transposed per batch
//               (V path: row=token -> Vt[b][col][token%2048], ldc/sC describe Vt)
template <bool CAUSAL_K, bool SKIP_UPPER, int OUT_MODE>
__global__ __launch_bounds__(256) void gemm_bf16x3(
    const u16* __restrict__ Ah, const u16* __restrict__ Al,
    const u16* __restrict__ Bh, const u16* __restrict__ Bl,
    float* __restrict__ Cf, u16* __restrict__ Ch, u16* __restrict__ Cl,
    const float* __restrict__ bias, int K, int lda, int ldb, int ldc, float alpha,
    long long sA, long long sB, long long sC) {
  const int m0 = blockIdx.y * 64;
  const int n0 = blockIdx.x * 128;
  if (SKIP_UPPER && n0 > m0 + 63) return;  // fully masked; softmax zeroes it

  const long long z = blockIdx.z;
  Ah += z * sA; Al += z * sA;
  Bh += z * sB; Bl += z * sB;

  __shared__ u16 AsH[2][64 * LP], AsL[2][64 * LP];
  __shared__ u16 BsH[2][128 * LP], BsL[2][128 * LP];

  const int tid  = threadIdx.x;
  const int lane = tid & 31;
  const int wave = tid >> 5;
  const int wm   = (wave & 3) * 16;
  const int wn   = (wave >> 2) * 64;
  const int lr   = lane & 15;
  const int h    = lane >> 4;  // lane-half -> K sub-range per ISA 16-bit layouts

  v8f acc[4];
#pragma unroll
  for (int j = 0; j < 4; j++) acc[j] = (v8f){0.f, 0.f, 0.f, 0.f, 0.f, 0.f, 0.f, 0.f};

  const int kEnd = CAUSAL_K ? ((K < m0 + 64) ? K : (m0 + 64)) : K;
  const int nT = kEnd >> 5;  // 32-deep k-chunks

  // Prologue: post tile 0 into buffer 0 (TDM; wave 0 only, uniform operands).
  if (wave == 0) {
    const long long ao = (long long)m0 * lda;
    const long long bo = (long long)n0 * ldb;
    tdm_tile_2d(Ah + ao, lda, 64, lds_off(AsH[0]));
    tdm_tile_2d(Al + ao, lda, 64, lds_off(AsL[0]));
    tdm_tile_2d(Bh + bo, ldb, 128, lds_off(BsH[0]));
    tdm_tile_2d(Bl + bo, ldb, 128, lds_off(BsL[0]));
  }

  for (int i = 0; i < nT; i++) {
    const int buf = i & 1;
    if (wave == 0) {
      if (i + 1 < nT) {  // post next tile into the other buffer
        const int nb = buf ^ 1;
        const long long k1 = (long long)(i + 1) * 32;
        const long long ao = (long long)m0 * lda + k1;
        const long long bo = (long long)n0 * ldb + k1;
        tdm_tile_2d(Ah + ao, lda, 64, lds_off(AsH[nb]));
        tdm_tile_2d(Al + ao, lda, 64, lds_off(AsL[nb]));
        tdm_tile_2d(Bh + bo, ldb, 128, lds_off(BsH[nb]));
        tdm_tile_2d(Bl + bo, ldb, 128, lds_off(BsL[nb]));
        __builtin_amdgcn_s_wait_tensorcnt(4);  // oldest 4 (tile i) landed
      } else {
        __builtin_amdgcn_s_wait_tensorcnt(0);
      }
    }
    __syncthreads();  // tile i visible to all waves

    // A fragment (16x32): lane(m=lr, half h) holds K runs [8h,+8) and [16+8h,+8)
    const u16* arow  = &AsH[buf][(wm + lr) * LP + h * 8];
    const u16* arowl = &AsL[buf][(wm + lr) * LP + h * 8];
    v16bf a_hi = frag_ld(arow, arow + 16);
    v16bf a_lo = frag_ld(arowl, arowl + 16);
#pragma unroll
    for (int j = 0; j < 4; j++) {
      // B fragment (32x16): lane(n=lr, half h) holds K runs [8h,+8) and [16+8h,+8)
      const u16* brow  = &BsH[buf][(wn + j * 16 + lr) * LP + h * 8];
      const u16* browl = &BsL[buf][(wn + j * 16 + lr) * LP + h * 8];
      v16bf b_hi = frag_ld(brow, brow + 16);
      v16bf b_lo = frag_ld(browl, browl + 16);
      acc[j] = __builtin_amdgcn_wmma_f32_16x16x32_bf16(false, a_hi, false, b_hi,
                                                       (short)0, acc[j], false, false);
      acc[j] = __builtin_amdgcn_wmma_f32_16x16x32_bf16(false, a_hi, false, b_lo,
                                                       (short)0, acc[j], false, false);
      acc[j] = __builtin_amdgcn_wmma_f32_16x16x32_bf16(false, a_lo, false, b_hi,
                                                       (short)0, acc[j], false, false);
    }
    __syncthreads();  // all done reading buf -> safe for TDM overwrite
  }

  // ---- epilogue: C/D layout VGPR i -> rows i and i+8 ----
  const long long zc = z * sC;
  const int rbase = m0 + wm + h * 8;
#pragma unroll
  for (int j = 0; j < 4; j++) {
    const int col = n0 + wn + j * 16 + lr;
    const float bv = bias ? bias[col] : 0.f;
#pragma unroll
    for (int i = 0; i < 8; i++) {
      const int row = rbase + i;
      const float val = acc[j][i] * alpha + bv;
      if (OUT_MODE == 0) {
        Cf[zc + (long long)row * ldc + col] = val;
      } else if (OUT_MODE == 1) {
        const long long idx = zc + (long long)row * ldc + col;
        const u16 hh = bf16_rne(val);
        Ch[idx] = hh;
        Cl[idx] = bf16_rne(val - bf16_to_f(hh));
      } else {
        // transposed per-batch store: token rows are 2048 per batch
        const long long idx =
            (long long)(row >> 11) * sC + (long long)col * ldc + (row & 2047);
        const u16 hh = bf16_rne(val);
        Ch[idx] = hh;
        Cl[idx] = bf16_rne(val - bf16_to_f(hh));
      }
    }
  }
}

// Causal row softmax: reads fp32 scores[B][S][S], writes P as hi/lo bf16.
__global__ __launch_bounds__(256) void softmax_causal(const float* __restrict__ S,
                                                      u16* __restrict__ Ph,
                                                      u16* __restrict__ Pl, int n) {
  const int r = blockIdx.x;
  const long long ro = ((long long)blockIdx.y * n + r) * (long long)n;
  const float* row = S + ro;
  const int tid  = threadIdx.x;
  const int lane = tid & 31;
  const int wave = tid >> 5;
  __shared__ float red[8];

  float vals[8];
  float m = -INFINITY;
#pragma unroll
  for (int j = 0; j < 8; j++) {
    const int c = tid + j * 256;
    const float v = (c <= r) ? row[c] : -INFINITY;
    vals[j] = v;
    m = fmaxf(m, v);
  }
#pragma unroll
  for (int off = 16; off > 0; off >>= 1) m = fmaxf(m, __shfl_xor(m, off, 32));
  if (lane == 0) red[wave] = m;
  __syncthreads();
  m = red[0];
#pragma unroll
  for (int w = 1; w < 8; w++) m = fmaxf(m, red[w]);
  __syncthreads();

  float e[8];
  float s = 0.f;
#pragma unroll
  for (int j = 0; j < 8; j++) {
    e[j] = __expf(vals[j] - m);
    s += e[j];
  }
#pragma unroll
  for (int off = 16; off > 0; off >>= 1) s += __shfl_xor(s, off, 32);
  if (lane == 0) red[wave] = s;
  __syncthreads();
  s = red[0];
#pragma unroll
  for (int w = 1; w < 8; w++) s += red[w];
  const float inv = 1.0f / s;
#pragma unroll
  for (int j = 0; j < 8; j++) {
    const float p = e[j] * inv;
    const u16 hh = bf16_rne(p);
    Ph[ro + tid + j * 256] = hh;
    Pl[ro + tid + j * 256] = bf16_rne(p - bf16_to_f(hh));
  }
}

extern "C" void kernel_launch(void* const* d_in, const int* in_sizes, int n_in,
                              void* d_out, int out_size, void* d_ws, size_t ws_size,
                              hipStream_t stream) {
  (void)in_sizes; (void)n_in; (void)out_size; (void)ws_size;
  constexpr int B = 4, S = 2048, D = 1024;
  constexpr long long MD = (long long)B * S * D;  // 8,388,608
  constexpr long long WN = (long long)D * D;      // 1,048,576
  constexpr long long SSB = (long long)B * S * S; // 16,777,216

  const float* x  = (const float*)d_in[0];
  const float* Wq = (const float*)d_in[1];
  const float* bq = (const float*)d_in[2];
  const float* Wk = (const float*)d_in[3];
  const float* bk = (const float*)d_in[4];
  const float* Wv = (const float*)d_in[5];
  const float* bv = (const float*)d_in[6];
  const float* Wo = (const float*)d_in[7];
  const float* bo = (const float*)d_in[8];
  float* out = (float*)d_out;

  char* p = (char*)d_ws;
  float* Sc = (float*)p;  p += SSB * 4;  // fp32 scores
  u16* xh  = (u16*)p;     p += MD * 2;
  u16* xl  = (u16*)p;     p += MD * 2;
  u16* Wqh = (u16*)p;     p += WN * 2;   // transposed [N,K] hi/lo
  u16* Wql = (u16*)p;     p += WN * 2;
  u16* Wkh = (u16*)p;     p += WN * 2;
  u16* Wkl = (u16*)p;     p += WN * 2;
  u16* Wvh = (u16*)p;     p += WN * 2;
  u16* Wvl = (u16*)p;     p += WN * 2;
  u16* Woh = (u16*)p;     p += WN * 2;
  u16* Wol = (u16*)p;     p += WN * 2;
  u16* Qh  = (u16*)p;     p += MD * 2;
  u16* Ql  = (u16*)p;     p += MD * 2;
  u16* Kh  = (u16*)p;     p += MD * 2;
  u16* Kl  = (u16*)p;     p += MD * 2;
  u16* Vth = (u16*)p;     p += MD * 2;   // V transposed: [B][D][S]
  u16* Vtl = (u16*)p;     p += MD * 2;
  u16* Ph  = (u16*)p;     p += SSB * 2;
  u16* Pl  = (u16*)p;     p += SSB * 2;
  u16* Ath = (u16*)p;     p += MD * 2;
  u16* Atl = (u16*)p;     p += MD * 2;

  const dim3 blk(256);
  const dim3 tblk(32, 8);
  const dim3 tgrd(32, 32);

  // 0) operand prep
  split_bf16<<<dim3(MD / 1024), blk, 0, stream>>>(x, xh, xl, MD);
  transpose_split<<<tgrd, tblk, 0, stream>>>(Wq, Wqh, Wql);
  transpose_split<<<tgrd, tblk, 0, stream>>>(Wk, Wkh, Wkl);
  transpose_split<<<tgrd, tblk, 0, stream>>>(Wv, Wvh, Wvl);
  transpose_split<<<tgrd, tblk, 0, stream>>>(Wo, Woh, Wol);

  // 1) projections (B = W^T stored [N,K])
  const dim3 gProj(D / 128, (B * S) / 64, 1);
  gemm_bf16x3<false, false, 1><<<gProj, blk, 0, stream>>>(
      xh, xl, Wqh, Wql, nullptr, Qh, Ql, bq, D, D, D, D, 1.0f, 0, 0, 0);
  gemm_bf16x3<false, false, 1><<<gProj, blk, 0, stream>>>(
      xh, xl, Wkh, Wkl, nullptr, Kh, Kl, bk, D, D, D, D, 1.0f, 0, 0, 0);
  // V: store transposed per batch -> Vt[b][d][s] (ldc = S, sC = D*S)
  gemm_bf16x3<false, false, 2><<<gProj, blk, 0, stream>>>(
      xh, xl, Wvh, Wvl, nullptr, Vth, Vtl, bv, D, D, D, S, 1.0f, 0, 0,
      (long long)D * S);

  // 2) scores = Q @ K^T / sqrt(64) -> fp32 (B = K stored [S,D] == [N,K])
  const dim3 gScore(S / 128, S / 64, B);
  gemm_bf16x3<false, true, 0><<<gScore, blk, 0, stream>>>(
      Qh, Ql, Kh, Kl, Sc, nullptr, nullptr, nullptr, D, D, D, S, 0.125f,
      (long long)S * D, (long long)S * D, (long long)S * S);

  // 3) causal softmax -> P hi/lo bf16
  softmax_causal<<<dim3(S, B), blk, 0, stream>>>(Sc, Ph, Pl, S);

  // 4) attn = P @ V (B = Vt stored [D,S] == [N,K]), causal k-limit
  const dim3 gAV(D / 128, S / 64, B);
  gemm_bf16x3<true, false, 1><<<gAV, blk, 0, stream>>>(
      Ph, Pl, Vth, Vtl, nullptr, Ath, Atl, nullptr, S, S, S, D, 1.0f,
      (long long)S * S, (long long)D * S, (long long)S * D);

  // 5) out = attn @ Wo + bo -> fp32
  gemm_bf16x3<false, false, 0><<<gProj, blk, 0, stream>>>(
      Ath, Atl, Woh, Wol, out, nullptr, nullptr, bo, D, D, D, D, 1.0f, 0, 0, 0);
}
